// Decoder_37821482008625
// MI455X (gfx1250) — compile-verified, matrix operand order checked by reading, not run on hardware
//
#include <hip/hip_runtime.h>

// ---------------------------------------------------------------------------
// MI455X (gfx1250) LSTM-decoder:
//   prep:       fp32 weights -> bf16, transposed to [N][K] (K contiguous),
//               z_cat = [z1|z2] -> bf16
//   lstm_main:  persistent WG per 32 batch rows; 20 steps of 2-layer LSTM,
//               all matmuls via v_wmma_f32_16x16x32_bf16, states in LDS/regs,
//               gate nonlinearities via native v_tanh_f32 (gfx1250)
//   proj:       hs @ Wmu / Wlv (WMMA) + reparameterize elementwise
// ---------------------------------------------------------------------------

typedef __bf16 bf16;
typedef __attribute__((ext_vector_type(16))) __bf16 v16bf;
typedef __attribute__((ext_vector_type(8)))  float  v8f;
typedef __attribute__((ext_vector_type(4)))  int    i32x4;

#define BS   8192
#define TT   20
#define FF   80
#define ZZ   32
#define HH   256
#define N4H  1024

// workspace layout (bf16 elements)
#define OFF_W1T  0u                      // [1024][64]
#define OFF_U1T  65536u                  // [1024][256]
#define OFF_W2T  327680u                 // [1024][256]
#define OFF_U2T  589824u                 // [1024][256]
#define OFF_WMUT 851968u                 // [80][256]
#define OFF_WLVT 872448u                 // [80][256]
#define OFF_ZC   892928u                 // [8192][64]
#define WS_ELEMS 1417216u

union FragU { v16bf v; i32x4 q[2]; };

__device__ __forceinline__ v16bf load_frag(const bf16* p) {
  // elements 0..7  = K kb..kb+7, elements 8..15 = K kb+16..kb+23
  FragU f;
  f.q[0] = *(const i32x4*)(p);
  f.q[1] = *(const i32x4*)(p + 16);
  return f.v;
}

__device__ __forceinline__ v8f wmma_bf16(v16bf a, v16bf b, v8f c) {
  return __builtin_amdgcn_wmma_f32_16x16x32_bf16(false, a, false, b,
                                                 (short)0, c, false, false);
}

// gfx1250 native v_tanh_f32 when available; exp-based fallback otherwise
#if __has_builtin(__builtin_amdgcn_tanhf)
__device__ __forceinline__ float tanh_f(float x) {
  return __builtin_amdgcn_tanhf(x);
}
#else
__device__ __forceinline__ float tanh_f(float x) {
  float e = __expf(2.0f * x);
  return (e - 1.0f) / (e + 1.0f);
}
#endif

__device__ __forceinline__ float sigm(float x) {
  // sigmoid(x) = 0.5*tanh(0.5x) + 0.5  -> v_mul + v_tanh + v_fma
  return __builtin_fmaf(0.5f, tanh_f(0.5f * x), 0.5f);
}

// --------------------------- prep kernel -----------------------------------
__global__ __launch_bounds__(256) void prep_kernel(
    const float* __restrict__ W1, const float* __restrict__ U1,
    const float* __restrict__ W2, const float* __restrict__ U2,
    const float* __restrict__ Wmu, const float* __restrict__ Wlv,
    const float* __restrict__ z1, const float* __restrict__ z2,
    bf16* __restrict__ ws)
{
  unsigned id = blockIdx.x * 256u + threadIdx.x;
  if (id < OFF_U1T) {                                  // W1t [n][k] n<1024 k<64
    unsigned f = id - OFF_W1T, n = f >> 6, k = f & 63u;
    ws[id] = (bf16)W1[k * N4H + n];
  } else if (id < OFF_W2T) {                           // U1t
    unsigned f = id - OFF_U1T, n = f >> 8, k = f & 255u;
    ws[id] = (bf16)U1[k * N4H + n];
  } else if (id < OFF_U2T) {                           // W2t
    unsigned f = id - OFF_W2T, n = f >> 8, k = f & 255u;
    ws[id] = (bf16)W2[k * N4H + n];
  } else if (id < OFF_WMUT) {                          // U2t
    unsigned f = id - OFF_U2T, n = f >> 8, k = f & 255u;
    ws[id] = (bf16)U2[k * N4H + n];
  } else if (id < OFF_WLVT) {                          // Wmut [n][k] n<80
    unsigned f = id - OFF_WMUT, n = f >> 8, k = f & 255u;
    ws[id] = (bf16)Wmu[k * FF + n];
  } else if (id < OFF_ZC) {                            // Wlvt
    unsigned f = id - OFF_WLVT, n = f >> 8, k = f & 255u;
    ws[id] = (bf16)Wlv[k * FF + n];
  } else if (id < WS_ELEMS) {                          // zcat [b][64]
    unsigned f = id - OFF_ZC, b = f >> 6, j = f & 63u;
    ws[id] = (bf16)((j < 32u) ? z1[b * ZZ + j] : z2[b * ZZ + (j - 32u)]);
  }
}

// --------------------------- main LSTM kernel ------------------------------
__global__ __launch_bounds__(256) void lstm_main(
    const bf16* __restrict__ zcat, const bf16* __restrict__ W1t,
    const bf16* __restrict__ U1t,  const bf16* __restrict__ W2t,
    const bf16* __restrict__ U2t,  const float* __restrict__ b1,
    const float* __restrict__ b2,  float* __restrict__ out_hs)
{
  __shared__ bf16 h1s[32][264];   // [m][h], +8 pad vs LDS banks
  __shared__ bf16 h2s[32][264];
  __shared__ bf16 zs[32][72];     // z_cat tile

  const int tid   = threadIdx.x;
  const int wave  = tid >> 5;
  const int lane  = tid & 31;
  const int lhalf = lane >> 4;     // 0 | 1 : K-half selector per WMMA layout
  const int lmod  = lane & 15;
  const int h0    = wave * 32;     // this wave owns h columns [h0, h0+32)
  const int b0    = blockIdx.x * 32;

  for (int i = tid; i < 32 * 64; i += 256) {
    int m = i >> 6, j = i & 63;
    zs[m][j] = zcat[(size_t)(b0 + m) * 64 + j];
  }
  for (int i = tid; i < 32 * 256; i += 256) {
    int m = i >> 8, j = i & 255;
    h1s[m][j] = (bf16)0.0f;
    h2s[m][j] = (bf16)0.0f;
  }

  // cell state in registers: fixed lane<->element map matching C/D layout
  float c1r[2][2][8], c2r[2][2][8];
#pragma unroll
  for (int a = 0; a < 2; ++a)
#pragma unroll
    for (int m = 0; m < 2; ++m)
#pragma unroll
      for (int r = 0; r < 8; ++r) { c1r[a][m][r] = 0.0f; c2r[a][m][r] = 0.0f; }

  float bb1[2][4], bb2[2][4];
#pragma unroll
  for (int ht = 0; ht < 2; ++ht)
#pragma unroll
    for (int g = 0; g < 4; ++g) {
      int n = g * HH + h0 + ht * 16 + lmod;
      bb1[ht][g] = b1[n];
      bb2[ht][g] = b2[n];
    }

  __syncthreads();

#pragma unroll 1
  for (int t = 0; t < TT; ++t) {
    v8f acc[2][2][4];

    // ================= layer 1: z = b1 + zcat@W1 + h1@U1 =================
#pragma unroll
    for (int ht = 0; ht < 2; ++ht)
#pragma unroll
      for (int mt = 0; mt < 2; ++mt)
#pragma unroll
        for (int g = 0; g < 4; ++g) {
          v8f a;
#pragma unroll
          for (int r = 0; r < 8; ++r) a[r] = bb1[ht][g];
          acc[ht][mt][g] = a;
        }

    // K over z_cat (K=64)
#pragma unroll
    for (int kk = 0; kk < 2; ++kk) {
      const int kb = kk * 32 + lhalf * 8;
#pragma unroll
      for (int ht = 0; ht < 2; ++ht) {
        v16bf bfr[4];
#pragma unroll
        for (int g = 0; g < 4; ++g)
          bfr[g] = load_frag(W1t + (size_t)(g * HH + h0 + ht * 16 + lmod) * 64 + kb);
#pragma unroll
        for (int mt = 0; mt < 2; ++mt) {
          v16bf af = load_frag(&zs[mt * 16 + lmod][kb]);
#pragma unroll
          for (int g = 0; g < 4; ++g)
            acc[ht][mt][g] = wmma_bf16(af, bfr[g], acc[ht][mt][g]);
        }
      }
    }
    // K over h1 (K=256)
#pragma unroll 2
    for (int kk = 0; kk < 8; ++kk) {
      const int kb = kk * 32 + lhalf * 8;
#pragma unroll
      for (int ht = 0; ht < 2; ++ht) {
        v16bf bfr[4];
#pragma unroll
        for (int g = 0; g < 4; ++g)
          bfr[g] = load_frag(U1t + (size_t)(g * HH + h0 + ht * 16 + lmod) * HH + kb);
#pragma unroll
        for (int mt = 0; mt < 2; ++mt) {
          v16bf af = load_frag(&h1s[mt * 16 + lmod][kb]);
#pragma unroll
          for (int g = 0; g < 4; ++g)
            acc[ht][mt][g] = wmma_bf16(af, bfr[g], acc[ht][mt][g]);
        }
      }
    }
    __syncthreads();   // all reads of h1s done before anyone rewrites it

    // gates layer 1
#pragma unroll
    for (int ht = 0; ht < 2; ++ht) {
      const int hc = h0 + ht * 16 + lmod;
#pragma unroll
      for (int mt = 0; mt < 2; ++mt)
#pragma unroll
        for (int r = 0; r < 8; ++r) {
          float zi = acc[ht][mt][0][r];
          float zf = acc[ht][mt][1][r];
          float zg = acc[ht][mt][2][r];
          float zo = acc[ht][mt][3][r];
          float c  = sigm(zf) * c1r[ht][mt][r] + sigm(zi) * tanh_f(zg);
          c1r[ht][mt][r] = c;
          float h  = sigm(zo) * tanh_f(c);
          h1s[mt * 16 + r + lhalf * 8][hc] = (bf16)h;
        }
    }
    __syncthreads();   // new h1 visible

    // ================= layer 2: z = b2 + h1@W2 + h2@U2 =================
#pragma unroll
    for (int ht = 0; ht < 2; ++ht)
#pragma unroll
      for (int mt = 0; mt < 2; ++mt)
#pragma unroll
        for (int g = 0; g < 4; ++g) {
          v8f a;
#pragma unroll
          for (int r = 0; r < 8; ++r) a[r] = bb2[ht][g];
          acc[ht][mt][g] = a;
        }

#pragma unroll 2
    for (int kk = 0; kk < 8; ++kk) {
      const int kb = kk * 32 + lhalf * 8;
#pragma unroll
      for (int ht = 0; ht < 2; ++ht) {
        v16bf bfr[4];
#pragma unroll
        for (int g = 0; g < 4; ++g)
          bfr[g] = load_frag(W2t + (size_t)(g * HH + h0 + ht * 16 + lmod) * HH + kb);
#pragma unroll
        for (int mt = 0; mt < 2; ++mt) {
          v16bf af = load_frag(&h1s[mt * 16 + lmod][kb]);
#pragma unroll
          for (int g = 0; g < 4; ++g)
            acc[ht][mt][g] = wmma_bf16(af, bfr[g], acc[ht][mt][g]);
        }
      }
    }
#pragma unroll 2
    for (int kk = 0; kk < 8; ++kk) {
      const int kb = kk * 32 + lhalf * 8;
#pragma unroll
      for (int ht = 0; ht < 2; ++ht) {
        v16bf bfr[4];
#pragma unroll
        for (int g = 0; g < 4; ++g)
          bfr[g] = load_frag(U2t + (size_t)(g * HH + h0 + ht * 16 + lmod) * HH + kb);
#pragma unroll
        for (int mt = 0; mt < 2; ++mt) {
          v16bf af = load_frag(&h2s[mt * 16 + lmod][kb]);
#pragma unroll
          for (int g = 0; g < 4; ++g)
            acc[ht][mt][g] = wmma_bf16(af, bfr[g], acc[ht][mt][g]);
        }
      }
    }
    __syncthreads();   // all reads of h2s done

    // gates layer 2 (+ emit hs output)
#pragma unroll
    for (int ht = 0; ht < 2; ++ht) {
      const int hc = h0 + ht * 16 + lmod;
#pragma unroll
      for (int mt = 0; mt < 2; ++mt)
#pragma unroll
        for (int r = 0; r < 8; ++r) {
          float zi = acc[ht][mt][0][r];
          float zf = acc[ht][mt][1][r];
          float zg = acc[ht][mt][2][r];
          float zo = acc[ht][mt][3][r];
          float c  = sigm(zf) * c2r[ht][mt][r] + sigm(zi) * tanh_f(zg);
          c2r[ht][mt][r] = c;
          float h  = sigm(zo) * tanh_f(c);
          int mrow = mt * 16 + r + lhalf * 8;
          h2s[mrow][hc] = (bf16)h;
          out_hs[(((size_t)(b0 + mrow)) * TT + t) * HH + hc] = h;
        }
    }
    __syncthreads();   // new h2 visible for next step
  }
}

// --------------------------- projection kernel -----------------------------
__global__ __launch_bounds__(256) void proj_kernel(
    const float* __restrict__ hs, const bf16* __restrict__ Wmut,
    const bf16* __restrict__ Wlvt, const float* __restrict__ bmu,
    const float* __restrict__ blv, const float* __restrict__ eps,
    float* __restrict__ out_mu, float* __restrict__ out_lv,
    float* __restrict__ out_s)
{
  const int wave  = threadIdx.x >> 5;
  const int lane  = threadIdx.x & 31;
  const int lhalf = lane >> 4;
  const int lmod  = lane & 15;
  const size_t rowbase = ((size_t)blockIdx.x * 8 + wave) * 16;

  v8f amu[5], alv[5];
#pragma unroll
  for (int nt = 0; nt < 5; ++nt) {
    float bm = bmu[nt * 16 + lmod];
    float bl = blv[nt * 16 + lmod];
#pragma unroll
    for (int r = 0; r < 8; ++r) { amu[nt][r] = bm; alv[nt][r] = bl; }
  }

#pragma unroll 2
  for (int kk = 0; kk < 8; ++kk) {
    const int kb = kk * 32 + lhalf * 8;
    const float* ap = hs + (rowbase + lmod) * HH + kb;
    v16bf af;
#pragma unroll
    for (int e = 0; e < 8; ++e) af[e] = (bf16)ap[e];
#pragma unroll
    for (int e = 0; e < 8; ++e) af[8 + e] = (bf16)ap[16 + e];
#pragma unroll
    for (int nt = 0; nt < 5; ++nt) {
      v16bf bm = load_frag(Wmut + (size_t)(nt * 16 + lmod) * HH + kb);
      v16bf bl = load_frag(Wlvt + (size_t)(nt * 16 + lmod) * HH + kb);
      amu[nt] = wmma_bf16(af, bm, amu[nt]);
      alv[nt] = wmma_bf16(af, bl, alv[nt]);
    }
  }

#pragma unroll
  for (int nt = 0; nt < 5; ++nt)
#pragma unroll
    for (int r = 0; r < 8; ++r) {
      size_t m  = rowbase + r + lhalf * 8;
      int    n  = nt * 16 + lmod;
      size_t ix = m * FF + n;
      float mu = amu[nt][r];
      float lv = alv[nt][r];
      out_mu[ix] = mu;
      out_lv[ix] = lv;
      // reference swaps args: sample = eps * exp(0.5*mu) + lv
      out_s[ix] = eps[ix] * __expf(0.5f * mu) + lv;
    }
}

// --------------------------- launcher --------------------------------------
extern "C" void kernel_launch(void* const* d_in, const int* in_sizes, int n_in,
                              void* d_out, int out_size, void* d_ws, size_t ws_size,
                              hipStream_t stream) {
  (void)in_sizes; (void)n_in; (void)out_size; (void)ws_size;
  const float* z1  = (const float*)d_in[1];
  const float* z2  = (const float*)d_in[2];
  const float* eps = (const float*)d_in[3];
  const float* W1  = (const float*)d_in[4];
  const float* U1  = (const float*)d_in[5];
  const float* b1  = (const float*)d_in[6];
  const float* W2  = (const float*)d_in[7];
  const float* U2  = (const float*)d_in[8];
  const float* b2  = (const float*)d_in[9];
  const float* Wmu = (const float*)d_in[10];
  const float* bmu = (const float*)d_in[11];
  const float* Wlv = (const float*)d_in[12];
  const float* blv = (const float*)d_in[13];

  bf16* ws   = (bf16*)d_ws;
  bf16* W1t  = ws + OFF_W1T;
  bf16* U1t  = ws + OFF_U1T;
  bf16* W2t  = ws + OFF_W2T;
  bf16* U2t  = ws + OFF_U2T;
  bf16* Wmut = ws + OFF_WMUT;
  bf16* Wlvt = ws + OFF_WLVT;
  bf16* zc   = ws + OFF_ZC;

  float* out    = (float*)d_out;
  float* out_hs = out;                                  // [8192,20,256]
  float* out_mu = out_hs + (size_t)BS * TT * HH;        // [8192,20,80]
  float* out_lv = out_mu + (size_t)BS * TT * FF;
  float* out_s  = out_lv + (size_t)BS * TT * FF;

  prep_kernel<<<(WS_ELEMS + 255u) / 256u, 256, 0, stream>>>(
      W1, U1, W2, U2, Wmu, Wlv, z1, z2, ws);
  lstm_main<<<BS / 32, 256, 0, stream>>>(
      zc, W1t, U1t, W2t, U2t, b1, b2, out_hs);
  proj_kernel<<<(BS * TT) / 128, 256, 0, stream>>>(
      out_hs, Wmut, Wlvt, bmu, blv, eps, out_mu, out_lv, out_s);
}
